// EdgeHeatmapGenerator_6167573037181
// MI455X (gfx1250) — matrix operand order repaired and checked by reading, not compiled, not run on hardware
//
#include <hip/hip_runtime.h>

// ---------------------------------------------------------------------------
// EdgeHeatmapGenerator for MI455X (gfx1250, wave32, WMMA)
//   B=32, E=16384, N=1000, D=128
//   out[b,i,j] = 0.5*(heat + heat^T), heat[b,src,dst] = sigmoid(mlp(edge))*10
// f16 WMMA (16x16x32) dense layers with hoisted A-fragments, rcp-based SiLU,
// packed f16 converts, butterfly head, fused symmetric scatter, NT zero-fill.
// ---------------------------------------------------------------------------

typedef _Float16 v16h __attribute__((ext_vector_type(16)));
typedef _Float16 v8h  __attribute__((ext_vector_type(8)));
typedef _Float16 v4h  __attribute__((ext_vector_type(4)));
typedef _Float16 v2h  __attribute__((ext_vector_type(2)));
typedef float    v8f  __attribute__((ext_vector_type(8)));
typedef float    v4f  __attribute__((ext_vector_type(4)));

#define DD      128       // feature dim
#define TILE_E  128       // edge rows per block
#define E_CNT   16384
#define N_CNT   1000
#define B_CNT   32

// ---------------------------------------------------------------------------
// Pass 1: zero the (B,N,N) output with non-temporal 128-bit stores.
// ---------------------------------------------------------------------------
__global__ __launch_bounds__(256)
void zero_out_kernel(v4f* __restrict__ out, long n4) {
    long i      = (long)blockIdx.x * blockDim.x + threadIdx.x;
    long stride = (long)gridDim.x * blockDim.x;
    v4f z = {0.0f, 0.0f, 0.0f, 0.0f};
    for (; i < n4; i += stride)
        __builtin_nontemporal_store(z, out + i);
}

// Packed f32x4 -> f16x4 using v_cvt_pk_rtz_f16_f32 (2 ops instead of 4+pack).
__device__ __forceinline__ v4h cvt4(float4 w) {
    v2h lo = __builtin_bit_cast(v2h, __builtin_amdgcn_cvt_pkrtz(w.x, w.y));
    v2h hi = __builtin_bit_cast(v2h, __builtin_amdgcn_cvt_pkrtz(w.z, w.w));
    return __builtin_shufflevector(lo, hi, 0, 1, 2, 3);
}

// A-fragment (16x32 f16, M = lane%16), contiguous K to match B:
//   lanes 0-15 : K = kBase + {0..15}
//   lanes 16-31: K = kBase + {16..31}
__device__ __forceinline__ v16h load_a_frag(const _Float16* sM, int m, int kBase, int lane) {
    int koff = kBase + ((lane & 16) ? 16 : 0);
    v8h lo = *(const v8h*)(sM + m * DD + koff);
    v8h hi = *(const v8h*)(sM + m * DD + koff + 8);
    return __builtin_shufflevector(lo, hi, 0,1,2,3,4,5,6,7,8,9,10,11,12,13,14,15);
}

// SiLU with single-instruction v_rcp_f32 instead of IEEE divide chain.
__device__ __forceinline__ float fast_silu(float x) {
    return x * __builtin_amdgcn_rcpf(1.0f + __expf(-x));
}

// ---------------------------------------------------------------------------
// Pass 2: fused MLP + symmetric scatter. 256 threads = 8 waves, 128 rows/block.
// ---------------------------------------------------------------------------
__global__ __launch_bounds__(256)
void edge_mlp_kernel(const float* __restrict__ edge_attr,
                     const int*   __restrict__ edge_index,
                     const float* __restrict__ W1, const float* __restrict__ b1,
                     const float* __restrict__ W2, const float* __restrict__ b2,
                     const float* __restrict__ Wo, const float* __restrict__ bo,
                     float* __restrict__ out) {
    extern __shared__ __align__(128) char smem[];
    _Float16* sW1 = (_Float16*)smem;            // 128x128 f16 (32KB)
    _Float16* sW2 = sW1 + DD * DD;              // 32KB
    _Float16* sA  = sW2 + DD * DD;              // activations tile (32KB)
    _Float16* sH  = sA  + DD * DD;              // layer-1 output (32KB)
    float*    sB1 = (float*)(sH + DD * DD);     // biases / head weights
    float*    sB2 = sB1 + DD;
    float*    sWo = sB2 + DD;
    float*    sV  = sWo + DD;                   // TILE_E edge values

    const int  tid  = threadIdx.x;
    const int  lane = tid & 31;
    const int  wave = tid >> 5;                 // 0..7
    const long row0 = (long)blockIdx.x * TILE_E;
    const int  b    = (int)(row0 / E_CNT);      // E divisible by TILE_E
    const int  e0   = (int)(row0 % E_CNT);

    // --- stage weights f32 -> f16 into LDS (L2-resident, cheap) -------------
    const float4* W1v = (const float4*)W1;
    const float4* W2v = (const float4*)W2;
    for (int i = tid; i < DD * DD / 4; i += 256) {
        ((v4h*)sW1)[i] = cvt4(W1v[i]);
        ((v4h*)sW2)[i] = cvt4(W2v[i]);
    }
    // --- stage edge_attr tile f32 -> f16 ------------------------------------
    const float4* Av = (const float4*)(edge_attr + row0 * DD);
    for (int i = tid; i < TILE_E * DD / 4; i += 256)
        ((v4h*)sA)[i] = cvt4(Av[i]);

    if (tid < DD) { sB1[tid] = b1[tid]; sB2[tid] = b2[tid]; sWo[tid] = Wo[tid]; }

    // speculative prefetch of the next block's tile into L2
    if (tid == 0 && (row0 + TILE_E) < (long)B_CNT * E_CNT)
        __builtin_prefetch(edge_attr + (row0 + TILE_E) * DD, 0, 1);

    __syncthreads();

    const int m0   = wave * 16;                 // wave's 16 rows
    const int mRow = m0 + (lane & 15);          // A-fragment row for this lane
    const int mloc = (lane & 16) ? 8 : 0;       // C-layout row offset
    const int nlo  = lane & 15;
    const int bsel = (lane & 16) ? 16 : 0;      // B-fragment K-half select

    // ----------------------- layer 1: h = silu(A W1^T + b1) -----------------
    {
        // Hoist the 4 K-chunk A-fragments: invariant across all 8 N-tiles.
        v16h a0 = load_a_frag(sA, mRow,  0, lane);
        v16h a1 = load_a_frag(sA, mRow, 32, lane);
        v16h a2 = load_a_frag(sA, mRow, 64, lane);
        v16h a3 = load_a_frag(sA, mRow, 96, lane);
        #pragma unroll
        for (int nT = 0; nT < 8; ++nT) {
            const _Float16* wrow = sW1 + (nT * 16 + nlo) * DD + bsel;
            v8f acc = {};
            acc = __builtin_amdgcn_wmma_f32_16x16x32_f16(false, a0, false,
                    *(const v16h*)(wrow +  0), (short)0, acc, false, false);
            acc = __builtin_amdgcn_wmma_f32_16x16x32_f16(false, a1, false,
                    *(const v16h*)(wrow + 32), (short)0, acc, false, false);
            acc = __builtin_amdgcn_wmma_f32_16x16x32_f16(false, a2, false,
                    *(const v16h*)(wrow + 64), (short)0, acc, false, false);
            acc = __builtin_amdgcn_wmma_f32_16x16x32_f16(false, a3, false,
                    *(const v16h*)(wrow + 96), (short)0, acc, false, false);
            const int   n    = nT * 16 + nlo;
            const float bias = sB1[n];
            #pragma unroll
            for (int r = 0; r < 8; ++r) {
                float s = fast_silu(acc[r] + bias);
                sH[(m0 + mloc + r) * DD + n] = (_Float16)s;
            }
        }
    }
    __syncthreads();

    // ------------- layer 2 + head: p = silu(H W2^T + b2) . Wo ---------------
    float p[8] = {0.f, 0.f, 0.f, 0.f, 0.f, 0.f, 0.f, 0.f};
    {
        v16h a0 = load_a_frag(sH, mRow,  0, lane);
        v16h a1 = load_a_frag(sH, mRow, 32, lane);
        v16h a2 = load_a_frag(sH, mRow, 64, lane);
        v16h a3 = load_a_frag(sH, mRow, 96, lane);
        #pragma unroll
        for (int nT = 0; nT < 8; ++nT) {
            const _Float16* wrow = sW2 + (nT * 16 + nlo) * DD + bsel;
            v8f acc = {};
            acc = __builtin_amdgcn_wmma_f32_16x16x32_f16(false, a0, false,
                    *(const v16h*)(wrow +  0), (short)0, acc, false, false);
            acc = __builtin_amdgcn_wmma_f32_16x16x32_f16(false, a1, false,
                    *(const v16h*)(wrow + 32), (short)0, acc, false, false);
            acc = __builtin_amdgcn_wmma_f32_16x16x32_f16(false, a2, false,
                    *(const v16h*)(wrow + 64), (short)0, acc, false, false);
            acc = __builtin_amdgcn_wmma_f32_16x16x32_f16(false, a3, false,
                    *(const v16h*)(wrow + 96), (short)0, acc, false, false);
            const int   n    = nT * 16 + nlo;
            const float bias = sB2[n];
            const float w    = sWo[n];
            #pragma unroll
            for (int r = 0; r < 8; ++r)
                p[r] += fast_silu(acc[r] + bias) * w;
        }
    }

    // butterfly-reduce the dot product across the 16 lanes sharing a row
    const float bo_v = *bo;
    #pragma unroll
    for (int r = 0; r < 8; ++r) {
        float v = p[r];
        v += __shfl_xor(v, 8, 16);
        v += __shfl_xor(v, 4, 16);
        v += __shfl_xor(v, 2, 16);
        v += __shfl_xor(v, 1, 16);
        if (nlo == 0) {
            float z = v + bo_v;
            // sigmoid(z)*10 with single v_rcp_f32
            sV[m0 + mloc + r] = 10.0f * __builtin_amdgcn_rcpf(1.0f + __expf(-z));
        }
    }
    __syncthreads();

    // ------------------- fused symmetric scatter ----------------------------
    if (tid < TILE_E) {
        const int e   = e0 + tid;
        const int src = edge_index[(long)b * 2 * E_CNT + e];
        const int dst = edge_index[(long)b * 2 * E_CNT + E_CNT + e];
        const float v = 0.5f * sV[tid];
        float* outb = out + (long)b * N_CNT * N_CNT;
        atomicAdd(outb + (long)src * N_CNT + dst, v);
        atomicAdd(outb + (long)dst * N_CNT + src, v);
    }
}

// ---------------------------------------------------------------------------
extern "C" void kernel_launch(void* const* d_in, const int* in_sizes, int n_in,
                              void* d_out, int out_size, void* d_ws, size_t ws_size,
                              hipStream_t stream) {
    const float* edge_attr  = (const float*)d_in[0];
    const int*   edge_index = (const int*)d_in[1];
    const float* W1 = (const float*)d_in[2];
    const float* b1 = (const float*)d_in[3];
    const float* W2 = (const float*)d_in[4];
    const float* b2 = (const float*)d_in[5];
    const float* Wo = (const float*)d_in[6];
    const float* bo = (const float*)d_in[7];
    float* out = (float*)d_out;

    // zero the (B,N,N) output: 32M floats = 8M float4
    const long n4 = ((long)B_CNT * N_CNT * N_CNT) / 4;
    zero_out_kernel<<<2048, 256, 0, stream>>>((v4f*)out, n4);

    // fused MLP + scatter: 4096 blocks of 256 threads, 128 edges each
    const int nblocks = (B_CNT * E_CNT) / TILE_E;
    const size_t smem = (size_t)4 * DD * DD * sizeof(_Float16)   // W1,W2,A,H
                      + (size_t)(3 * DD + TILE_E) * sizeof(float);
    edge_mlp_kernel<<<nblocks, 256, smem, stream>>>(edge_attr, edge_index,
                                                    W1, b1, W2, b2, Wo, bo, out);
}
